// CustomCRF_55774445305979
// MI455X (gfx1250) — matrix operand order repaired and checked by reading, not compiled, not run on hardware
//
#include <hip/hip_runtime.h>

// CRF forward (log partition) for T=16384, N=1024 on MI455X (gfx1250).
//
// logsumexp recurrence factorized into a linear bf16 matvec:
//   E[n,p] = exp(trans[n,p] - rowmax(trans)[n])   (precomputed once, 2 MB bf16)
//   v[p]   = exp(a[p] - max(a))                   (per step)
//   a'[n]  = mT[n] + ma + log( (E @ v)[n] ) + feat[n]
// E is partitioned across 16 workgroups, 64 rows each, resident in 128 KB LDS
// (staged by the Tensor Data Mover: tensor_load_to_lds + s_wait_tensorcnt).
// E@v is computed with v_wmma_f32_16x16x32_bf16 (B = v broadcast over columns).
// Cross-WG step sync: monotonic global atomic counter (release/acquire).

#define T_STEPS   16384
#define NTAGS     1024
#define START_TAG 1022
#define STOP_TAG  1023
#define NEG_INF_F (-10000.0f)
#define NWG       16
#define ROWS_WG   (NTAGS / NWG)   // 64
#define BLOCK     256             // 8 wave32

typedef __attribute__((ext_vector_type(16))) __bf16 v16bf;
typedef __attribute__((ext_vector_type(8)))  float  v8f;
typedef __attribute__((ext_vector_type(4)))  unsigned int tdm_v4u;
typedef __attribute__((ext_vector_type(8)))  int          tdm_v8i;
typedef __attribute__((ext_vector_type(4)))  int          tdm_v4i;

union ABPack { uint4 u[2]; v16bf v; };

__device__ __forceinline__ unsigned short f2bf(float f) {
  union { float f; unsigned u; } c; c.f = f;
  return (unsigned short)((c.u + 0x7FFFu + ((c.u >> 16) & 1u)) >> 16);
}

// ---------------- prep: E = exp(trans - rowmax), mT = rowmax, init a/sync ----
__global__ void __launch_bounds__(BLOCK)
crf_prep(const float* __restrict__ trans,
         unsigned short* __restrict__ E,
         float* __restrict__ mT,
         float* __restrict__ aBuf,
         unsigned int* __restrict__ syncCnt) {
  __shared__ float red[8];
  const int row  = blockIdx.x;
  const int tid  = threadIdx.x;
  const int lane = tid & 31;
  const int wave = tid >> 5;

  const float4 t4 = ((const float4*)(trans + (size_t)row * NTAGS))[tid];
  float m = fmaxf(fmaxf(t4.x, t4.y), fmaxf(t4.z, t4.w));
#pragma unroll
  for (int off = 16; off > 0; off >>= 1) m = fmaxf(m, __shfl_xor(m, off, 32));
  if (lane == 0) red[wave] = m;
  __syncthreads();
  float rm = red[0];
#pragma unroll
  for (int i = 1; i < 8; ++i) rm = fmaxf(rm, red[i]);
  if (tid == 0) mT[row] = rm;

  const unsigned p0 = (unsigned)f2bf(__expf(t4.x - rm)) |
                      ((unsigned)f2bf(__expf(t4.y - rm)) << 16);
  const unsigned p1 = (unsigned)f2bf(__expf(t4.z - rm)) |
                      ((unsigned)f2bf(__expf(t4.w - rm)) << 16);
  unsigned* Er = (unsigned*)(E + (size_t)row * NTAGS);
  Er[tid * 2 + 0] = p0;
  Er[tid * 2 + 1] = p1;

  if (row == 0) {
    for (int i = tid; i < NTAGS; i += BLOCK)
      aBuf[i] = (i == START_TAG) ? 0.0f : NEG_INF_F;
    if (tid == 0) *syncCnt = 0u;
  }
}

// ---------------- main: 16 persistent WGs, one E slice each ------------------
__global__ void __launch_bounds__(BLOCK)
crf_forward(const float* __restrict__ decoded,
            const float* __restrict__ trans,
            const unsigned short* __restrict__ E,
            const float* __restrict__ mT,
            float* __restrict__ aBuf,          // [2][NTAGS] ping-pong
            unsigned int* __restrict__ syncCnt,
            float* __restrict__ out) {
  extern __shared__ unsigned char smem[];
  unsigned short* sE   = (unsigned short*)smem;                  // 131072 B
  unsigned short* sV   = (unsigned short*)(smem + 131072);       //   2048 B
  float*          sMT  = (float*)(smem + 133120);                //    256 B
  float*          sPart= (float*)(smem + 133376);                //    512 B
  float*          sRed = (float*)(smem + 133888);                //   1024 B

  const int wg   = blockIdx.x;
  const int tid  = threadIdx.x;
  const int lane = tid & 31;
  const int wave = tid >> 5;
  const int r0   = wg * ROWS_WG;

  // ---- Stage this WG's 64x1024 bf16 E slice into LDS (128 KB). ------------
#if defined(__gfx1250__) && __has_builtin(__builtin_amdgcn_tensor_load_to_lds)
  // Tensor Data Mover: one 2-D descriptor moves the whole tile, tracked by
  // TENSORcnt. D# layout per CDNA5 ISA ch.8 (group0 128b, group1 256b).
  if (wave == 0) {
    const unsigned long long gaddr =
        (unsigned long long)(const void*)(E + (size_t)r0 * NTAGS);
    tdm_v4u g0;
    g0[0] = 1u;                                   // count=1 (valid descriptor)
    g0[1] = 0u;                                   // lds_addr = 0 (sE base)
    g0[2] = (unsigned)(gaddr & 0xFFFFFFFFull);    // global_addr[31:0]
    g0[3] = (unsigned)((gaddr >> 32) & 0x1FFFFFFull) | (2u << 30); // type=2
    tdm_v8i g1;
    g1[0] = 0x10000;                // data_size=2B (code 1), wg_mask=0
    g1[1] = (int)(0x0400u << 16);   // tensor_dim0[15:0]=1024 in [31:16]
    g1[2] = (int)(0x0400u << 16);   // tensor_dim0 hi=0 | tensor_dim1 lo=1024
    g1[3] = (int)(0x0400u << 16);   // tensor_dim1 hi=0 | tile_dim0=1024
    g1[4] = ROWS_WG;                // tile_dim1=64, tile_dim2=0
    g1[5] = NTAGS;                  // tensor_dim0_stride[31:0]=1024
    g1[6] = 0;                      // stride hi / tensor_dim1_stride lo
    g1[7] = 0;                      // tensor_dim1_stride hi
    tdm_v4i g2 = {0, 0, 0, 0};      // 2-D tensor: higher dims unused
    tdm_v4i g3 = {0, 0, 0, 0};
#if __clang_major__ >= 23
    tdm_v8i gx = {0, 0, 0, 0, 0, 0, 0, 0};   // extra operand (clang-23 form)
    __builtin_amdgcn_tensor_load_to_lds(g0, g1, g2, g3, gx, 0);
#else
    __builtin_amdgcn_tensor_load_to_lds(g0, g1, g2, g3, 0);
#endif
    __builtin_amdgcn_s_wait_tensorcnt(0);
  }
  for (int i = tid; i < ROWS_WG; i += BLOCK) sMT[i] = mT[r0 + i];
#else
  {
    const uint4* src = (const uint4*)(E + (size_t)r0 * NTAGS);
    uint4* dst = (uint4*)sE;
    for (int i = tid; i < (ROWS_WG * NTAGS) / 8; i += BLOCK) dst[i] = src[i];
    for (int i = tid; i < ROWS_WG; i += BLOCK) sMT[i] = mT[r0 + i];
  }
#endif
  __syncthreads();

  // Per-wave WMMA tile assignment: 4 row-tiles (16 rows) x 2 K-halves (512).
  const int rt   = wave & 3;
  const int kh   = wave >> 2;
  const int arow = 16 * rt + (lane & 15);
  const int aoff = (lane < 16) ? 0 : 8;    // ISA 16-bit A layout K offset
  const int boff = (lane < 16) ? 0 : 8;    // K offset matching A layout
  const unsigned short* eRow  = sE + (size_t)arow * NTAGS + kh * 512 + aoff;
  const unsigned short* vHalf = sV + kh * 512 + boff;

  for (int s = 0; s < T_STEPS; ++s) {
    const float* aCur = aBuf + ((s & 1) ? NTAGS : 0);
    float*       aNxt = aBuf + ((s & 1) ? 0 : NTAGS);

    // -------- phase A: max(a), v = exp(a - ma) as bf16 in LDS --------------
    const float4 a4 = ((const float4*)aCur)[tid];
    float m = fmaxf(fmaxf(a4.x, a4.y), fmaxf(a4.z, a4.w));
#pragma unroll
    for (int off = 16; off > 0; off >>= 1) m = fmaxf(m, __shfl_xor(m, off, 32));
    if (lane == 0) sRed[wave] = m;
    __syncthreads();
    float ma = sRed[0];
#pragma unroll
    for (int i = 1; i < 8; ++i) ma = fmaxf(ma, sRed[i]);
    const unsigned q0 = (unsigned)f2bf(__expf(a4.x - ma)) |
                        ((unsigned)f2bf(__expf(a4.y - ma)) << 16);
    const unsigned q1 = (unsigned)f2bf(__expf(a4.z - ma)) |
                        ((unsigned)f2bf(__expf(a4.w - ma)) << 16);
    ((unsigned*)sV)[tid * 2 + 0] = q0;
    ((unsigned*)sV)[tid * 2 + 1] = q1;
    __syncthreads();

    // -------- phase B: w = E_slice @ v via 16 bf16 WMMAs per wave ----------
    v8f acc = {};
#pragma unroll
    for (int c = 0; c < 16; ++c) {
      ABPack A, B;
      A.u[0] = *(const uint4*)(eRow + 32 * c);        // K +0..7  (rel. aoff)
      A.u[1] = *(const uint4*)(eRow + 32 * c + 16);   // K +16..23
      B.u[0] = *(const uint4*)(vHalf + 32 * c);       // v broadcast over N
      B.u[1] = *(const uint4*)(vHalf + 32 * c + 16);
      acc = __builtin_amdgcn_wmma_f32_16x16x32_bf16(
          false, A.v, false, B.v, (short)0, acc, false, false);
    }
    // B columns identical => every lane holds the result column.
    if (lane == 0 || lane == 16) {
      const int rbase = 16 * rt + ((lane >> 4) << 3); // +0 or +8
#pragma unroll
      for (int r = 0; r < 8; ++r)
        sPart[kh * ROWS_WG + rbase + r] = acc[r];
    }
    __syncthreads();

    // -------- phase C: combine halves, log, emission, store a_next ---------
    if (tid < ROWS_WG) {
      const float wsum = sPart[tid] + sPart[ROWS_WG + tid];
      const float na = sMT[tid] + ma + __logf(wsum) +
                       decoded[(size_t)s * NTAGS + r0 + tid];
      aNxt[r0 + tid] = na;
    } else if (tid < 64 + 4) {
      const int sp = s + 8;  // prefetch emissions 8 steps ahead
      if (sp < T_STEPS)
        __builtin_prefetch(
            (const void*)(decoded + (size_t)sp * NTAGS + r0 + (tid - 64) * 16),
            0, 1);
    }
    __syncthreads();

    // -------- grid sync: monotonic counter over 16 co-resident WGs ---------
    if (tid == 0) {
      __threadfence();
      __hip_atomic_fetch_add(syncCnt, 1u, __ATOMIC_ACQ_REL,
                             __HIP_MEMORY_SCOPE_AGENT);
      const unsigned tgt = (unsigned)NWG * (unsigned)(s + 1);
      while (__hip_atomic_load(syncCnt, __ATOMIC_ACQUIRE,
                               __HIP_MEMORY_SCOPE_AGENT) < tgt)
        __builtin_amdgcn_s_sleep(2);
      __threadfence();
    }
    __syncthreads();
  }

  // -------- terminal: alpha = logsumexp(a_final + trans[STOP]) -------------
  if (wg == 0) {
    const float* aF = aBuf + ((T_STEPS & 1) ? NTAGS : 0);
    const float4 a4 = ((const float4*)aF)[tid];
    const float4 t4 =
        ((const float4*)(trans + (size_t)STOP_TAG * NTAGS))[tid];
    const float e0 = a4.x + t4.x, e1 = a4.y + t4.y;
    const float e2 = a4.z + t4.z, e3 = a4.w + t4.w;
    float m2 = fmaxf(fmaxf(e0, e1), fmaxf(e2, e3));
#pragma unroll
    for (int off = 16; off > 0; off >>= 1) m2 = fmaxf(m2, __shfl_xor(m2, off, 32));
    if (lane == 0) sRed[wave] = m2;
    __syncthreads();
    float mg = sRed[0];
#pragma unroll
    for (int i = 1; i < 8; ++i) mg = fmaxf(mg, sRed[i]);
    float ss = __expf(e0 - mg) + __expf(e1 - mg) +
               __expf(e2 - mg) + __expf(e3 - mg);
#pragma unroll
    for (int off = 16; off > 0; off >>= 1) ss += __shfl_xor(ss, off, 32);
    __syncthreads();
    if (lane == 0) sRed[wave] = ss;
    __syncthreads();
    if (tid == 0) {
      float tot = 0.0f;
#pragma unroll
      for (int i = 0; i < 8; ++i) tot += sRed[i];
      out[0] = mg + __logf(tot);
    }
  }
}

// ---------------------------------------------------------------------------
extern "C" void kernel_launch(void* const* d_in, const int* in_sizes, int n_in,
                              void* d_out, int out_size, void* d_ws,
                              size_t ws_size, hipStream_t stream) {
  (void)in_sizes; (void)n_in; (void)out_size; (void)ws_size;
  const float* decoded = (const float*)d_in[0];
  const float* trans   = (const float*)d_in[1];
  // d_in[2] (raw_outputs) and d_in[3] (outputs) are unused by the reference.
  float* out = (float*)d_out;

  char* ws = (char*)d_ws;
  unsigned short* E    = (unsigned short*)ws;                         // 2 MB
  float*          mT   = (float*)(ws + 2u * 1024u * 1024u);           // 4 KB
  float*          aBuf = (float*)(ws + 2u * 1024u * 1024u + 4096u);   // 8 KB
  unsigned int*   sync =
      (unsigned int*)(ws + 2u * 1024u * 1024u + 4096u + 8192u);

  crf_prep<<<NTAGS, BLOCK, 0, stream>>>(trans, E, mT, aBuf, sync);

  const size_t smemBytes = 134912;  // 128K E + 2K v + mT/partials/reduce
  crf_forward<<<NWG, BLOCK, smemBytes, stream>>>(decoded, trans, E, mT, aBuf,
                                                 sync, out);
}
